// RootSpike_GLM_53669911331365
// MI455X (gfx1250) — matrix operand order repaired and checked by reading, not compiled, not run on hardware
//
#include <hip/hip_runtime.h>
#include <hip/hip_bf16.h>

typedef _Float16 v16h __attribute__((ext_vector_type(16)));
typedef float    v8f  __attribute__((ext_vector_type(8)));
typedef unsigned int u32x4 __attribute__((ext_vector_type(4)));
typedef int          i32x8 __attribute__((ext_vector_type(8)));
typedef int          i32x4 __attribute__((ext_vector_type(4)));

#define SUBN 20
#define CONO 13

__device__ __forceinline__ unsigned lds_addr_of(const void* p) {
    // generic pointer to LDS object: low 32 bits are the LDS byte address
    return (unsigned)(size_t)p;
}

// ---------------------------------------------------------------------------
// 1) raised-cosine log-time basis  [CO, TN]
// ---------------------------------------------------------------------------
__global__ void cos_basis_kernel(float* __restrict__ cosB, int TN, int CO) {
    int id = blockIdx.x * blockDim.x + threadIdx.x;
    if (id >= CO * TN) return;
    int i = id / TN, x = id - i * TN;
    float raw = 5.0f * logf((float)x + 1.0f + 1e-8f);
    float phi = 1.5707963267948966f * (float)i;
    const float PI = 3.14159265358979323846f;
    float b = 0.5f * cosf(raw - phi) + 0.5f;
    if (raw < phi - PI || raw > phi + PI) b = 0.f;
    cosB[id] = b;
}

// ---------------------------------------------------------------------------
// 2) out_filters rows: e_kern[SUB], i_kern[SUB], hist_kern[1]  -> d_out+T
// ---------------------------------------------------------------------------
__global__ void filters_kernel(const float* __restrict__ W,   // [SUB][CO][2]
                               const float* __restrict__ hw,  // [CO]
                               const float* __restrict__ cosB,
                               float* __restrict__ out, int TN, int CO, int SUB) {
    int id = blockIdx.x * blockDim.x + threadIdx.x;
    int total = (2 * SUB + 1) * TN;
    if (id >= total) return;
    int r = id / TN, k = id - r * TN;
    float acc = 0.f;
    if (r < SUB) {
        for (int i = 0; i < CO; ++i) acc = fmaf(W[(r * CO + i) * 2 + 0], cosB[i * TN + k], acc);
    } else if (r < 2 * SUB) {
        int s = r - SUB;
        for (int i = 0; i < CO; ++i) acc = fmaf(W[(s * CO + i) * 2 + 1], cosB[i * TN + k], acc);
    } else {
        for (int i = 0; i < CO; ++i) acc = fmaf(hw[i], cosB[i * TN + k], acc);
    }
    out[id] = acc;
}

// ---------------------------------------------------------------------------
// 3) gumbel-softmax routing: write f32 C_syn to d_out and padded f16 copy
//    Ch is [32][Kpad] half, rows >= SUBN and cols >= N zero-filled.
// ---------------------------------------------------------------------------
__global__ void routing_kernel(const float* __restrict__ raw, const float* __restrict__ u,
                               const float* __restrict__ tempP,
                               float* __restrict__ Cout, _Float16* __restrict__ Ch,
                               int N, int Kpad) {
    int col = blockIdx.x * blockDim.x + threadIdx.x;
    if (col >= Kpad) return;
    if (col >= N) {
        #pragma unroll
        for (int s = 0; s < 32; ++s) Ch[(size_t)s * Kpad + col] = (_Float16)0.f;
        return;
    }
    const float eps = 1e-7f;
    float temp = tempP[0];
    float l[SUBN];
    float m = -1e30f;
    #pragma unroll
    for (int s = 0; s < SUBN; ++s) {
        float g = -logf(-logf(u[(size_t)s * N + col] + eps) + eps);
        float v = (raw[(size_t)s * N + col] + g) / temp;
        l[s] = v;
        m = fmaxf(m, v);
    }
    float sum = 0.f;
    #pragma unroll
    for (int s = 0; s < SUBN; ++s) { l[s] = expf(l[s] - m); sum += l[s]; }
    float inv = 1.f / sum;
    #pragma unroll
    for (int s = 0; s < SUBN; ++s) {
        float p = l[s] * inv;
        Cout[(size_t)s * N + col] = p;
        Ch[(size_t)s * Kpad + col] = (_Float16)p;
    }
    #pragma unroll
    for (int s = SUBN; s < 32; ++s) Ch[(size_t)s * Kpad + col] = (_Float16)0.f;
}

// ---------------------------------------------------------------------------
// 4) WMMA GEMM:  outT[n][t] = sum_k S[t,k] * C[n,k]    (D stored transposed)
//    B matrix [32][Kpad] f16 is staged into LDS by the Tensor Data Mover
//    (one TENSOR_LOAD_TO_LDS per block, wave 0 issues, s_wait_tensorcnt).
//    One wave per 16-row M tile; two N tiles (subs 0-15, 16-31).
// ---------------------------------------------------------------------------
__global__ __launch_bounds__(256) void gemm_wmma_kernel(
        const float* __restrict__ S, const _Float16* __restrict__ Ch,
        float* __restrict__ outT, int T, int E, int Kpad, int nsub) {
    extern __shared__ _Float16 lB[];           // [32][Kpad] halves

    // ---- TDM: DMA the whole B panel (32 rows x Kpad*2 bytes) into LDS ----
    if (threadIdx.x < 32) {
        unsigned long long ga = (unsigned long long)(const void*)Ch;
        unsigned lds         = lds_addr_of(lB);
        unsigned rowB        = (unsigned)Kpad * 2u;          // bytes per row
        u32x4 g0;
        g0[0] = 1u;                                          // count=1, user D#
        g0[1] = lds;                                         // lds_addr
        g0[2] = (unsigned)(ga & 0xffffffffull);              // global_addr[31:0]
        g0[3] = (unsigned)((ga >> 32) & 0x1ffffffull)        // global_addr[56:32]
              | (2u << 30);                                  // type = 2 (image)
        i32x8 g1;
        g1[0] = 0;                                           // wg_mask=0, data_size=1B
        g1[1] = (int)((rowB & 0xffffu) << 16);               // tensor_dim0[15:0]
        g1[2] = (int)(((rowB >> 16) & 0xffffu))              // tensor_dim0[31:16]
              | (32 << 16);                                  // tensor_dim1 = 32 rows
        g1[3] = (int)((rowB & 0xffffu) << 16);               // tile_dim0 = row bytes
        g1[4] = 32;                                          // tile_dim1 = 32 rows
        g1[5] = (int)rowB;                                   // tensor_dim0_stride[31:0]
        g1[6] = 0;
        g1[7] = 0;
        i32x4 z4 = {};
        i32x8 z8 = {};
        __builtin_amdgcn_tensor_load_to_lds(g0, g1, z4, z4, z8, 0);
        __builtin_amdgcn_s_wait_tensorcnt(0);
    }
    __syncthreads();

    int wave = (blockIdx.x * blockDim.x + threadIdx.x) >> 5;
    int lane = threadIdx.x & 31;
    int m0 = wave * 16;
    if (m0 >= T) return;                       // wave-uniform: EXEC stays full
    int lm = lane & 15;
    int hi = lane >> 4;
    const float*    srow = S  + (size_t)(m0 + lm) * E;
    const _Float16* b0p  = lB + (size_t)lm * Kpad;
    const _Float16* b1p  = lB + (size_t)(lm + 16) * Kpad;
    v8f acc0 = {};
    v8f acc1 = {};
    int kfull = (E >> 5) << 5;
    int kt = 0;
    for (; kt < kfull; kt += 32) {
        int ka = kt + hi * 16;
        v16h a;
        #pragma unroll
        for (int j = 0; j < 8; ++j) a[j]     = (_Float16)srow[ka + j];
        #pragma unroll
        for (int j = 0; j < 8; ++j) a[8 + j] = (_Float16)srow[ka + 8 + j];
        __builtin_prefetch(srow + ka + 256, 0, 1);      // global_prefetch_b8
        int kb = kt + hi * 16;
        v16h b0 = *(const v16h*)(b0p + kb);             // ds_load_b128 x2
        v16h b1 = *(const v16h*)(b1p + kb);
        acc0 = __builtin_amdgcn_wmma_f32_16x16x32_f16(false, a, false, b0,
                                                      (short)0, acc0, false, false);
        acc1 = __builtin_amdgcn_wmma_f32_16x16x32_f16(false, a, false, b1,
                                                      (short)0, acc1, false, false);
    }
    for (; kt < Kpad; kt += 32) {              // guarded K tail (E not mult of 32)
        int ka = kt + hi * 16;
        v16h a;
        #pragma unroll
        for (int j = 0; j < 8; ++j)
            a[j]     = (ka + j      < E) ? (_Float16)srow[ka + j]      : (_Float16)0.f;
        #pragma unroll
        for (int j = 0; j < 8; ++j)
            a[8 + j] = (ka + 8 + j < E) ? (_Float16)srow[ka + 8 + j] : (_Float16)0.f;
        int kb = kt + hi * 16;
        v16h b0 = *(const v16h*)(b0p + kb);
        v16h b1 = *(const v16h*)(b1p + kb);
        acc0 = __builtin_amdgcn_wmma_f32_16x16x32_f16(false, a, false, b0,
                                                      (short)0, acc0, false, false);
        acc1 = __builtin_amdgcn_wmma_f32_16x16x32_f16(false, a, false, b1,
                                                      (short)0, acc1, false, false);
    }
    int tb = m0 + hi * 8;
    #pragma unroll
    for (int r = 0; r < 8; ++r)
        outT[(size_t)lm * T + tb + r] = acc0[r];
    if (16 + lm < nsub) {
        #pragma unroll
        for (int r = 0; r < 8; ++r)
            outT[(size_t)(16 + lm) * T + tb + r] = acc1[r];
    }
}

// ---------------------------------------------------------------------------
// 5) depthwise causal conv over channel-major syn:
//    synS[c][t] = sum_k ek[c][k]*synE[c][t-k] + ik[c][k]*synI[c][t-k]
//    x windows staged global->LDS with async loads (ASYNCcnt path).
// ---------------------------------------------------------------------------
__global__ __launch_bounds__(256) void dwconv_kernel(
        const float* __restrict__ xeT, const float* __restrict__ xiT,
        const float* __restrict__ ekAll, const float* __restrict__ ikAll,
        float* __restrict__ outT, int T, int TN) {
    extern __shared__ float sm[];
    float* ek = sm;                  // TN
    float* ik = sm + TN;             // TN
    float* xe = sm + 2 * TN;         // 256 + TN - 1
    float* xi = xe + (256 + TN - 1); // 256 + TN - 1
    int c   = blockIdx.y;
    int t0  = blockIdx.x * 256;
    int tid = threadIdx.x;
    for (int j = tid; j < TN; j += 256) {
        ek[j] = ekAll[(size_t)c * TN + j];
        ik[j] = ikAll[(size_t)c * TN + j];
    }
    int WL = 256 + TN - 1;
    for (int j = tid; j < WL; j += 256) {
        int ti = t0 - (TN - 1) + j;
        if (ti >= 0 && ti < T) {
            unsigned le = lds_addr_of(&xe[j]);
            unsigned li = lds_addr_of(&xi[j]);
            unsigned long long ge = (unsigned long long)(const void*)&xeT[(size_t)c * T + ti];
            unsigned long long gi = (unsigned long long)(const void*)&xiT[(size_t)c * T + ti];
            asm volatile("global_load_async_to_lds_b32 %0, %1, off"
                         :: "v"(le), "v"(ge) : "memory");
            asm volatile("global_load_async_to_lds_b32 %0, %1, off"
                         :: "v"(li), "v"(gi) : "memory");
        } else {
            xe[j] = 0.f;
            xi[j] = 0.f;
        }
    }
    asm volatile("s_wait_asynccnt 0x0" ::: "memory");
    __syncthreads();
    int t = t0 + tid;
    if (t >= T) return;
    float acc = 0.f;
    for (int k = 0; k < TN; ++k) {
        int j = tid + TN - 1 - k;
        acc = fmaf(ek[k], xe[j], acc);
        acc = fmaf(ik[k], xi[j], acc);
    }
    outT[(size_t)c * T + t] = acc;
}

// ---------------------------------------------------------------------------
// 6) spike-history filter + tree recurrence + sigmoid -> final_Z
// ---------------------------------------------------------------------------
__global__ __launch_bounds__(256) void tree_kernel(
        const float* __restrict__ Z, const float* __restrict__ hkG,
        const float* __restrict__ synT, const float* __restrict__ Wsub,
        const float* __restrict__ Theta, const int* __restrict__ Cden,
        float* __restrict__ outZ, int T, int TN) {
    extern __shared__ float sm2[];
    float* hk = sm2;        // TN
    float* zw = sm2 + TN;   // TN + 256
    int tid = threadIdx.x;
    int t0  = blockIdx.x * 256;
    for (int j = tid; j < TN; j += 256) hk[j] = hkG[j];
    int WL = TN + 255;
    for (int j = tid; j < WL; j += 256) {
        int zi = t0 - TN + j;
        zw[j] = (zi >= 0 && zi < T) ? Z[zi] : 0.f;
    }
    __syncthreads();
    int t = t0 + tid;
    if (t >= T) return;
    // hist[t] = sum_k hk[k] * Z[t-1-k]
    float hist = 0.f;
    for (int k = 0; k < TN; ++k)
        hist = fmaf(hk[k], zw[tid + TN - 1 - k], hist);
    float syn[SUBN], w2[SUBN], sOut[SUBN];
    #pragma unroll
    for (int c = 0; c < SUBN; ++c) {
        syn[c] = synT[(size_t)c * T + t];
        float w = Wsub[c];
        w2[c] = w * w;
        sOut[c] = 0.f;
    }
    #pragma unroll
    for (int j = SUBN - 1; j >= 1; --j) {
        float v = syn[j] + Theta[j];
        #pragma unroll
        for (int c = j + 1; c < SUBN; ++c)
            v = fmaf((float)Cden[j * SUBN + c] * w2[c], sOut[c], v);
        sOut[j] = tanhf(v);
    }
    float rin = hist + syn[0] + Theta[0];
    #pragma unroll
    for (int c = 1; c < SUBN; ++c)
        rin = fmaf((float)Cden[c] * w2[c], sOut[c], rin);
    outZ[t] = 1.f / (1.f + expf(-rin));
}

// ---------------------------------------------------------------------------
// launcher
// ---------------------------------------------------------------------------
static inline size_t alignup(size_t x) { return (x + 255) & ~(size_t)255; }

extern "C" void kernel_launch(void* const* d_in, const int* in_sizes, int n_in,
                              void* d_out, int out_size, void* d_ws, size_t ws_size,
                              hipStream_t stream) {
    const float* S_e   = (const float*)d_in[0];
    const float* S_i   = (const float*)d_in[1];
    const float* Z     = (const float*)d_in[2];
    const float* u_e   = (const float*)d_in[3];
    const float* u_i   = (const float*)d_in[4];
    const float* temp  = (const float*)d_in[5];
    const float* Wsyn  = (const float*)d_in[6];
    const float* histw = (const float*)d_in[7];
    const float* Wsub  = (const float*)d_in[8];
    const float* Theta = (const float*)d_in[9];
    const float* rawE  = (const float*)d_in[10];
    const float* rawI  = (const float*)d_in[11];
    const int*   Cden  = (const int*)d_in[12];

    const int T   = in_sizes[2];
    const int E   = in_sizes[0] / T;
    const int I   = in_sizes[1] / T;
    const int SUB = in_sizes[8];                 // 20
    const int CO  = in_sizes[7];                 // 13
    const int TN  = (out_size - T - SUB * (E + I)) / (2 * SUB + 1);  // 400
    const int KpE = ((E + 31) / 32) * 32;
    const int KpI = ((I + 31) / 32) * 32;

    char* ws = (char*)d_ws;
    size_t off = 0;
    float*    cosB = (float*)(ws + off);    off += alignup((size_t)CO * TN * sizeof(float));
    _Float16* ChE  = (_Float16*)(ws + off); off += alignup((size_t)32 * KpE * sizeof(_Float16));
    _Float16* ChI  = (_Float16*)(ws + off); off += alignup((size_t)32 * KpI * sizeof(_Float16));
    float*    synE = (float*)(ws + off);    off += alignup((size_t)SUB * T * sizeof(float));
    float*    synI = (float*)(ws + off);    off += alignup((size_t)SUB * T * sizeof(float));
    float*    synS = (float*)(ws + off);    off += alignup((size_t)SUB * T * sizeof(float));
    (void)ws_size; (void)n_in;

    float* outF    = (float*)d_out;
    float* finalZ  = outF;
    float* filters = outF + T;                     // [(2*SUB+1)][TN]
    float* CoutE   = outF + T + (2 * SUB + 1) * TN;
    float* CoutI   = CoutE + (size_t)SUB * E;

    cos_basis_kernel<<<(CO * TN + 255) / 256, 256, 0, stream>>>(cosB, TN, CO);

    filters_kernel<<<((2 * SUB + 1) * TN + 255) / 256, 256, 0, stream>>>(
        Wsyn, histw, cosB, filters, TN, CO, SUB);

    routing_kernel<<<(KpE + 255) / 256, 256, 0, stream>>>(rawE, u_e, temp, CoutE, ChE, E, KpE);
    routing_kernel<<<(KpI + 255) / 256, 256, 0, stream>>>(rawI, u_i, temp, CoutI, ChI, I, KpI);

    int tiles = (T + 15) / 16;
    int gblk  = (tiles + 7) / 8;   // 8 waves (M tiles) per 256-thread block
    size_t shE = (size_t)32 * KpE * sizeof(_Float16);
    size_t shI = (size_t)32 * KpI * sizeof(_Float16);
    gemm_wmma_kernel<<<gblk, 256, shE, stream>>>(S_e, ChE, synE, T, E, KpE, SUB);
    gemm_wmma_kernel<<<gblk, 256, shI, stream>>>(S_i, ChI, synI, T, I, KpI, SUB);

    dim3 cgrid((T + 255) / 256, SUB);
    size_t convSh = (size_t)(2 * TN + 2 * (256 + TN - 1)) * sizeof(float);
    dwconv_kernel<<<cgrid, 256, convSh, stream>>>(synE, synI, filters,
                                                  filters + (size_t)SUB * TN, synS, T, TN);

    size_t treeSh = (size_t)(2 * TN + 256) * sizeof(float);
    tree_kernel<<<(T + 255) / 256, 256, treeSh, stream>>>(
        Z, filters + (size_t)2 * SUB * TN, synS, Wsub, Theta, Cden, finalZ, T, TN);
}
